// SpatialAttentionLayer_87952340287888
// MI455X (gfx1250) — matrix operand order repaired and checked by reading, not compiled, not run on hardware
//
#include <hip/hip_runtime.h>
#include <hip/hip_bf16.h>

typedef __attribute__((ext_vector_type(16))) _Float16 v16h;
typedef __attribute__((ext_vector_type(8)))  _Float16 h8;
typedef __attribute__((ext_vector_type(8)))  float    v8f;
typedef __attribute__((ext_vector_type(4)))  float    f4;

#define NGROUP 84
#define NN     400
#define NPAD   416
#define NF     512
#define NH     64
#define LRELU_ALPHA 0.2f

__device__ __forceinline__ v8f wmma_f16(v16h a, v16h b, v8f c) {
  // D = A(16x32 f16) * B(32x16 f16) + C(16x16 f32)
  return __builtin_amdgcn_wmma_f32_16x16x32_f16(
      false, a, false, b, (short)0, c, false, false);
}

__device__ __forceinline__ v16h pack16(h8 lo, h8 hi) {
  v16h r;
#pragma unroll
  for (int e = 0; e < 8; ++e) { r[e] = lo[e]; r[e + 8] = hi[e]; }
  return r;
}

// ---------------------------------------------------------------- row sums
__global__ void k_rowsum(const float* __restrict__ features, float* __restrict__ rowsum) {
  const int lane = threadIdx.x & 31;
  const int wid  = threadIdx.x >> 5;
  const int idx  = blockIdx.x * 8 + wid;           // one wave per row
  const float* p = features + (size_t)idx * NF;
  float s = 0.f;
  for (int j = lane; j < NN; j += 32) s += p[j];
#pragma unroll
  for (int d = 16; d > 0; d >>= 1) s += __shfl_xor(s, d, 32);
  if (lane == 0) rowsum[idx] = s;
}

// ---------------------------------------------------------------- col sums
__global__ void k_colsum(const float* __restrict__ features, float* __restrict__ colsum) {
  const int g = blockIdx.y;
  const int j = blockIdx.x * 256 + threadIdx.x;
  if (j >= NN) return;
  const float* p = features + (size_t)g * NN * NF + j;
  float s = 0.f;
  for (int i = 0; i < NN; ++i) s += p[(size_t)i * NF];
  colsum[g * NN + j] = s;
}

// ------------------------------------ transpose + f32->f16: dst[c*rows+r] = src[r*cols+c]
__global__ void k_tr16(const float* __restrict__ src, _Float16* __restrict__ dst,
                       int rows, int cols) {
  const int c = blockIdx.x;
  for (int r = threadIdx.x; r < rows; r += blockDim.x)
    dst[(size_t)c * rows + r] = (_Float16)src[(size_t)r * cols + c];
}

// ------------------------------------------------- w1 = Wsa@a[:64], w2 = Wsa@a[64:]
__global__ void k_w12(const float* W0, const float* a0, const float* W1, const float* a1,
                      const float* W2, const float* a2, float* __restrict__ w12) {
  const int t = blockIdx.x;
  const float* Wsa = (t == 0) ? W0 : (t == 1) ? W1 : W2;
  const float* a   = (t == 0) ? a0 : (t == 1) ? a1 : a2;
  const int tid = threadIdx.x;           // 0..127
  const int which = tid >> 6, k = tid & 63;
  float s = 0.f;
  for (int c = 0; c < NH; ++c) s += Wsa[k * NH + c] * a[which * NH + c];
  w12[t * 128 + which * NH + k] = s;
}

// ------------------------------------------- feats = features @ W  (WMMA f16)
__global__ __launch_bounds__(128) void k_feats(const float* __restrict__ features,
                                               const _Float16* __restrict__ W16T,
                                               float* __restrict__ out,
                                               _Float16* __restrict__ feats16,
                                               _Float16* __restrict__ feats16T) {
  const int g  = blockIdx.y;
  const int tm = blockIdx.x;               // 0..25 (tile 25 = zero-pad rows 400..415)
  const int lane = threadIdx.x & 31;
  const int wid  = threadIdx.x >> 5;       // ntile 0..3 covers NH=64
  const int lr = lane & 15, lh = lane >> 4;
  const int col = wid * 16 + lr;

  if (tm == 25) {                          // block-uniform zero-pad path
    h8 z = {};
    const int rm = 400 + (threadIdx.x >> 3);
    const int cm = (threadIdx.x & 7) * 8;
    *(h8*)(feats16 + ((size_t)g * NPAD + rm) * NH + cm) = z;
    const int ct = threadIdx.x >> 1;
    const int rt = 400 + (threadIdx.x & 1) * 8;
    *(h8*)(feats16T + ((size_t)g * NH + ct) * NPAD + rt) = z;
    return;
  }

  const int row = tm * 16 + lr;            // < 400
  const float* arow = features + (size_t)(g * NN + row) * NF;
  v8f acc = {};
  for (int ks = 0; ks < NF / 32; ++ks) {
    const int kb = ks * 32 + lh * 8;
    f4 x0 = __builtin_nontemporal_load((const f4*)(arow + kb));
    f4 x1 = __builtin_nontemporal_load((const f4*)(arow + kb + 4));
    f4 y0 = __builtin_nontemporal_load((const f4*)(arow + kb + 16));
    f4 y1 = __builtin_nontemporal_load((const f4*)(arow + kb + 20));
    v16h a;
#pragma unroll
    for (int e = 0; e < 4; ++e) {
      a[e]      = (_Float16)x0[e];
      a[e + 4]  = (_Float16)x1[e];
      a[e + 8]  = (_Float16)y0[e];
      a[e + 12] = (_Float16)y1[e];
    }
    const _Float16* bp = W16T + (size_t)col * NF + ks * 32 + lh * 8;
    h8 b0 = *(const h8*)bp;
    h8 b1 = *(const h8*)(bp + 16);
    if (ks + 1 < NF / 32) __builtin_prefetch(arow + ks * 32 + 32, 0, 1);
    acc = wmma_f16(a, pack16(b0, b1), acc);
  }
  h8 t16;
#pragma unroll
  for (int v = 0; v < 8; ++v) t16[v] = (_Float16)acc[v];
  *(h8*)(feats16T + ((size_t)g * NH + col) * NPAD + tm * 16 + 8 * lh) = t16;
#pragma unroll
  for (int v = 0; v < 8; ++v) {
    const int m = tm * 16 + v + 8 * lh;    // < 400
    feats16[((size_t)g * NPAD + m) * NH + col] = t16[v];
    __builtin_nontemporal_store(acc[v], out + (size_t)(g * NN + m) * 256 + col);
  }
}

// ---- feat_out / feat_in / feat_geo : A synthesized on the fly, B = feats16T
__global__ __launch_bounds__(128) void k_featX(const float* __restrict__ features,
                                               const float* __restrict__ geo,
                                               const float* __restrict__ rowsum,
                                               const float* __restrict__ colsum,
                                               const _Float16* __restrict__ feats16T,
                                               _Float16* __restrict__ fo,
                                               _Float16* __restrict__ fi,
                                               _Float16* __restrict__ fg) {
  const int g = blockIdx.y, tm = blockIdx.x, t = blockIdx.z;
  const int lane = threadIdx.x & 31, wid = threadIdx.x >> 5;
  const int lr = lane & 15, lh = lane >> 4;
  const int col = wid * 16 + lr;
  _Float16* dst = (t == 0) ? fo : (t == 1) ? fi : fg;

  if (tm == 25) {                          // block-uniform zero-pad path
    h8 z = {};
    const int rm = 400 + (threadIdx.x >> 3);
    const int cm = (threadIdx.x & 7) * 8;
    *(h8*)(dst + ((size_t)g * NPAD + rm) * NH + cm) = z;
    return;
  }

  const int row = tm * 16 + lr;            // < 400
  float scale;
  if (t == 0)      scale = 1.f / (rowsum[g * NN + row] + 1.f);
  else if (t == 1) scale = 1.f / (colsum[g * NN + row] + 1.f);
  else             scale = 1.f;
  const float* arow = features + (size_t)(g * NN + row) * NF;

  v8f acc = {};
  for (int ks = 0; ks < 13; ++ks) {        // K = 400 -> 13 steps, tail via selects
    const int jb = ks * 32 + lh * 8;
    v16h a;
    if (t == 0) {
      // contiguous, in-bounds up to col 511 of the feature row
      f4 x0 = __builtin_nontemporal_load((const f4*)(arow + jb));
      f4 x1 = __builtin_nontemporal_load((const f4*)(arow + jb + 4));
      f4 y0 = __builtin_nontemporal_load((const f4*)(arow + jb + 16));
      f4 y1 = __builtin_nontemporal_load((const f4*)(arow + jb + 20));
#pragma unroll
      for (int e = 0; e < 4; ++e) {
        a[e]      = (_Float16)((jb + e      < NN ? x0[e] : 0.f) * scale);
        a[e + 4]  = (_Float16)((jb + 4 + e  < NN ? x1[e] : 0.f) * scale);
        a[e + 8]  = (_Float16)((jb + 16 + e < NN ? y0[e] : 0.f) * scale);
        a[e + 12] = (_Float16)((jb + 20 + e < NN ? y1[e] : 0.f) * scale);
      }
    } else if (t == 1) {
#pragma unroll
      for (int e = 0; e < 8; ++e) {
        const int ja = jb + e, jc = jb + 16 + e;
        const int jca = ja < NN ? ja : NN - 1;
        const int jcc = jc < NN ? jc : NN - 1;
        const float xa = features[(size_t)(g * NN + jca) * NF + row];
        const float xc = features[(size_t)(g * NN + jcc) * NF + row];
        a[e]     = (_Float16)(ja < NN ? xa * scale : 0.f);
        a[e + 8] = (_Float16)(jc < NN ? xc * scale : 0.f);
      }
    } else {
#pragma unroll
      for (int e = 0; e < 8; ++e) {
        const int ja = jb + e, jc = jb + 16 + e;
        const int jca = ja < NN ? ja : NN - 1;
        const int jcc = jc < NN ? jc : NN - 1;
        const float xa = geo[row * NN + jca];
        const float xc = geo[row * NN + jcc];
        a[e]     = (_Float16)(ja < NN ? xa : 0.f);
        a[e + 8] = (_Float16)(jc < NN ? xc : 0.f);
      }
    }
    const _Float16* bp = feats16T + ((size_t)g * NH + col) * NPAD + ks * 32 + lh * 8;
    h8 b0 = *(const h8*)bp;
    h8 b1 = *(const h8*)(bp + 16);
    acc = wmma_f16(a, pack16(b0, b1), acc);
  }
#pragma unroll
  for (int v = 0; v < 8; ++v) {
    const int m = tm * 16 + v + 8 * lh;    // < 400
    dst[((size_t)g * NPAD + m) * NH + col] = (_Float16)acc[v];
  }
}

// ---------------- attention: s1/s2, h2=featX@Wsa (LDS, transposed), softmax, o=att@h2
__global__ __launch_bounds__(256) void k_attn(const _Float16* __restrict__ feats16,
                                              const _Float16* __restrict__ fo,
                                              const _Float16* __restrict__ fi,
                                              const _Float16* __restrict__ fg,
                                              const _Float16* __restrict__ WsaT16,
                                              const float* __restrict__ w12,
                                              float* __restrict__ out) {
  __shared__ float s1[NN], s2[NPAD], rmax[NN], rinv[NN];
  __shared__ _Float16 h2T[NH * NPAD];      // transposed: [col][j]
  const int g = blockIdx.x, t = blockIdx.y;
  const _Float16* fx   = (t == 0) ? fo : (t == 1) ? fi : fg;
  const _Float16* WsaT = WsaT16 + (size_t)t * NH * NH;
  const float* w1 = w12 + t * 128;
  const float* w2 = w1 + NH;
  const int tid = threadIdx.x;

  // Phase A: s1[i] = feats[i]·w1, s2[j] = featX[j]·w2  (vectorized f16 reads)
  for (int i = tid; i < NN; i += 256) {
    const h8* p = (const h8*)(feats16 + ((size_t)g * NPAD + i) * NH);
    const h8* q = (const h8*)(fx      + ((size_t)g * NPAD + i) * NH);
    float a = 0.f, b = 0.f;
#pragma unroll
    for (int kk = 0; kk < 8; ++kk) {
      h8 pv = p[kk], qv = q[kk];
#pragma unroll
      for (int e = 0; e < 8; ++e) {
        a += (float)pv[e] * w1[kk * 8 + e];
        b += (float)qv[e] * w2[kk * 8 + e];
      }
    }
    s1[i] = a; s2[i] = b;
  }
  for (int i = tid; i < NPAD - NN; i += 256) s2[NN + i] = 0.f;
  __syncthreads();

  const int lane = tid & 31, wid = tid >> 5;
  const int lr = lane & 15, lh = lane >> 4;

  // Phase B: h2 = featX @ Wsa -> LDS transposed (padded rows come out zero)
  for (int job = wid; job < 26 * 4; job += 8) {
    const int tm = job >> 2, tn = job & 3;
    const int row = tm * 16 + lr;
    const int col = tn * 16 + lr;
    const _Float16* arow = fx + ((size_t)g * NPAD + row) * NH;
    v8f acc = {};
#pragma unroll
    for (int ks = 0; ks < 2; ++ks) {
      h8 a0 = *(const h8*)(arow + ks * 32 + lh * 8);
      h8 a1 = *(const h8*)(arow + ks * 32 + 16 + lh * 8);
      const _Float16* bp = WsaT + (size_t)col * NH + ks * 32 + lh * 8;
      h8 b0 = *(const h8*)bp;
      h8 b1 = *(const h8*)(bp + 16);
      acc = wmma_f16(pack16(a0, a1), pack16(b0, b1), acc);
    }
    h8 t16;
#pragma unroll
    for (int v = 0; v < 8; ++v) t16[v] = (_Float16)acc[v];
    *(h8*)(h2T + (size_t)col * NPAD + tm * 16 + 8 * lh) = t16;
  }
  __syncthreads();

  // Phase C: per-row softmax stats over j
  for (int i = tid; i < NN; i += 256) {
    const float si = s1[i];
    float mx = -3.0e38f;
    for (int j = 0; j < NN; ++j) {
      float x = si + s2[j];
      x = x > 0.f ? x : LRELU_ALPHA * x;
      mx = fmaxf(mx, x);
    }
    float den = 0.f;
    for (int j = 0; j < NN; ++j) {
      float x = si + s2[j];
      x = x > 0.f ? x : LRELU_ALPHA * x;
      den += __expf(x - mx);
    }
    rmax[i] = mx; rinv[i] = 1.f / den;
  }
  __syncthreads();

  // Phase D: o = att @ h2, att generated per-fragment (branchless, j>=400 -> 0)
  for (int job = wid; job < 25 * 4; job += 8) {
    const int tm = job >> 2, tn = job & 3;
    const int row = tm * 16 + lr;          // < 400
    const int col = tn * 16 + lr;
    const float si = s1[row], mi = rmax[row], ri = rinv[row];
    v8f acc = {};
    for (int ks = 0; ks < 13; ++ks) {
      const int jb = ks * 32 + lh * 8;
      v16h a;
#pragma unroll
      for (int e = 0; e < 8; ++e) {
        const int ja = jb + e, jc = jb + 16 + e;   // <= 415, s2 defined
        float x = si + s2[ja];
        x = x > 0.f ? x : LRELU_ALPHA * x;
        const float aa = __expf(x - mi) * ri;
        float y = si + s2[jc];
        y = y > 0.f ? y : LRELU_ALPHA * y;
        const float ac = __expf(y - mi) * ri;
        a[e]     = (_Float16)(ja < NN ? aa : 0.f);
        a[e + 8] = (_Float16)(jc < NN ? ac : 0.f);
      }
      const _Float16* bp = h2T + (size_t)col * NPAD + ks * 32 + lh * 8;
      h8 b0 = *(const h8*)bp;
      h8 b1 = *(const h8*)(bp + 16);
      acc = wmma_f16(a, pack16(b0, b1), acc);
    }
#pragma unroll
    for (int v = 0; v < 8; ++v) {
      const int m = tm * 16 + v + 8 * lh;  // < 400
      __builtin_nontemporal_store(acc[v],
          out + (size_t)(g * NN + m) * 256 + (1 + t) * NH + col);
    }
  }
}

extern "C" void kernel_launch(void* const* d_in, const int* in_sizes, int n_in,
                              void* d_out, int out_size, void* d_ws, size_t ws_size,
                              hipStream_t stream) {
  (void)in_sizes; (void)n_in; (void)out_size; (void)ws_size;
  const float* features = (const float*)d_in[0];
  const float* geo      = (const float*)d_in[1];
  const float* W        = (const float*)d_in[2];
  const float* W_out    = (const float*)d_in[3];
  const float* a_out    = (const float*)d_in[4];
  const float* W_in     = (const float*)d_in[5];
  const float* a_in     = (const float*)d_in[6];
  const float* W_geo    = (const float*)d_in[7];
  const float* a_geo    = (const float*)d_in[8];
  float* out = (float*)d_out;

  char* ws = (char*)d_ws;
  size_t off = 0;
  auto wsalloc = [&](size_t bytes) -> void* {
    void* p = ws + off;
    off += (bytes + 255) & ~(size_t)255;
    return p;
  };
  float* rowsum = (float*)wsalloc((size_t)NGROUP * NN * sizeof(float));
  float* colsum = (float*)wsalloc((size_t)NGROUP * NN * sizeof(float));
  float* w12    = (float*)wsalloc(3 * 128 * sizeof(float));
  _Float16* W16T     = (_Float16*)wsalloc((size_t)NF * NH * 2);
  _Float16* WsaT16   = (_Float16*)wsalloc((size_t)3 * NH * NH * 2);
  _Float16* feats16  = (_Float16*)wsalloc((size_t)NGROUP * NPAD * NH * 2);
  _Float16* feats16T = (_Float16*)wsalloc((size_t)NGROUP * NH * NPAD * 2);
  _Float16* fo       = (_Float16*)wsalloc((size_t)NGROUP * NPAD * NH * 2);
  _Float16* fi       = (_Float16*)wsalloc((size_t)NGROUP * NPAD * NH * 2);
  _Float16* fg       = (_Float16*)wsalloc((size_t)NGROUP * NPAD * NH * 2);

  k_rowsum<<<NGROUP * NN / 8, 256, 0, stream>>>(features, rowsum);
  k_colsum<<<dim3(2, NGROUP), 256, 0, stream>>>(features, colsum);
  k_w12<<<3, 128, 0, stream>>>(W_out, a_out, W_in, a_in, W_geo, a_geo, w12);
  k_tr16<<<NH, 128, 0, stream>>>(W, W16T, NF, NH);
  k_tr16<<<NH, 128, 0, stream>>>(W_out, WsaT16 + 0 * NH * NH, NH, NH);
  k_tr16<<<NH, 128, 0, stream>>>(W_in,  WsaT16 + 1 * NH * NH, NH, NH);
  k_tr16<<<NH, 128, 0, stream>>>(W_geo, WsaT16 + 2 * NH * NH, NH, NH);
  k_feats<<<dim3(26, NGROUP), 128, 0, stream>>>(features, W16T, out, feats16, feats16T);
  k_featX<<<dim3(26, NGROUP, 3), 128, 0, stream>>>(features, geo, rowsum, colsum,
                                                   feats16T, fo, fi, fg);
  k_attn<<<dim3(NGROUP, 3), 256, 0, stream>>>(feats16, fo, fi, fg,
                                              WsaT16, w12, out);
}
